// DeformableTransformerDecoderLayer_54992761258339
// MI455X (gfx1250) — matrix operand order repaired and checked by reading, not compiled, not run on hardware
//
#include <hip/hip_runtime.h>
#include <hip/hip_bf16.h>

// ---------------------------------------------------------------------------
// Problem constants (from reference)
// ---------------------------------------------------------------------------
#define C_BS   4
#define C_T    4
#define C_LQ   300
#define C_NQ   1200          // T*LQ tokens per batch
#define C_M1   4800          // BS*NQ
#define C_D    256
#define C_NH   8
#define C_DH   32
#define C_NF   4
#define C_NL   4
#define C_NP   4
#define C_S    5440          // 64*64+32*32+16*16+8*8
#define C_M2   87040         // BS*NF*S
#define C_DFFN 1024
#define ATTN_SCALE 0.17677669529663687f   // 1/sqrt(32)

typedef _Float16 half8_t __attribute__((ext_vector_type(8)));
typedef _Float16 v16h    __attribute__((ext_vector_type(16)));
typedef float    v8f     __attribute__((ext_vector_type(8)));
typedef int      v4i     __attribute__((ext_vector_type(4)));

// gfx1250 async global->LDS staging (guarded; falls back to reg + ds_store)
#if defined(__AMDGCN__) && __has_builtin(__builtin_amdgcn_global_load_async_to_lds_b128) && __has_builtin(__builtin_amdgcn_s_wait_asynccnt)
#define USE_ASYNC_LDS 1
#define GLOBAL_AS __attribute__((address_space(1)))
#define LDS_AS    __attribute__((address_space(3)))
#else
#define USE_ASYNC_LDS 0
#endif

// ---------------------------------------------------------------------------
// f32 -> f16 convert (grid stride)
// ---------------------------------------------------------------------------
__global__ void k_f32_to_f16(const float* __restrict__ src, _Float16* __restrict__ dst, int n) {
    for (int i = blockIdx.x * blockDim.x + threadIdx.x; i < n; i += gridDim.x * blockDim.x)
        dst[i] = (_Float16)src[i];
}

// q16 = f16(x + qp), x16 = f16(x)
__global__ void k_prep(const float* __restrict__ x, const float* __restrict__ qp,
                       _Float16* __restrict__ q16, _Float16* __restrict__ x16, int n) {
    int i = blockIdx.x * blockDim.x + threadIdx.x;
    if (i < n) {
        float xv = x[i];
        x16[i] = (_Float16)xv;
        q16[i] = (_Float16)(xv + qp[i]);
    }
}

// ---------------------------------------------------------------------------
// WMMA GEMM with async-LDS staged B panel.
//   C[M,N] = A[M,K](f16 row-major) * B[K,N](f16 row-major) + bias
// Block: 256 threads = 8 waves -> 128x64 output patch.
// Each wave: one 16-row strip x 4 N-subtiles (4 accumulators, A-frag reuse x4).
// B chunk (64x64 f16 = 8KB) double-buffered in LDS, staged with
// GLOBAL_LOAD_ASYNC_TO_LDS_B128, fenced by s_wait_asynccnt + barrier.
// ---------------------------------------------------------------------------
#define GEMM_BM 128
#define GEMM_BN 64
#define GEMM_KC 64
#define GEMM_RELU 1

__global__ void __launch_bounds__(256)
k_gemm_wmma(const _Float16* __restrict__ A, const _Float16* __restrict__ B,
            const float* __restrict__ bias,
            float* __restrict__ Cf, _Float16* __restrict__ Ch,
            int M, int K, int N, int flags) {
    __shared__ __align__(16) _Float16 lds[2 * GEMM_KC * GEMM_BN];   // 16 KB

    const int tid  = threadIdx.x;
    const int lane = tid & 31;
    const int wave = tid >> 5;
    const int nbn  = N / GEMM_BN;
    const int bm   = blockIdx.x / nbn;
    const int bn   = blockIdx.x - bm * nbn;
    const int mt   = M >> 4;

    int tm = bm * 8 + wave;
    const bool active = (tm < mt);
    if (!active) tm = mt - 1;                 // clamp: redundant compute, keeps barriers uniform

    // A fragment: lane L (row = L&15) holds K chunks [kb,kb+8) and [kb+16,kb+24)
    const int arow  = tm * 16 + (lane & 15);
    const int kb    = (lane >> 4) * 8;
    // B fragment: lane L holds column n = L&15, K rows [kb,kb+8) and [kb+16,kb+24)
    const int brow  = (lane & 15);
    const int bcol0 = bn * GEMM_BN;

    v8f acc0 = {}, acc1 = {}, acc2 = {}, acc3 = {};

    // stage one 64x64 B chunk into LDS buffer `buf` (512 x 16B, 2 per thread)
    auto stage = [&](int buf, int kk0) {
        #pragma unroll
        for (int i = 0; i < 2; ++i) {
            int t = tid + 256 * i;
            int r = t >> 3;
            int c = (t & 7) * 8;
            const _Float16* gp = B + (size_t)(kk0 + r) * N + bcol0 + c;
            _Float16* lp = &lds[buf * (GEMM_KC * GEMM_BN) + r * GEMM_BN + c];
#if USE_ASYNC_LDS
            __builtin_amdgcn_global_load_async_to_lds_b128(
                (GLOBAL_AS v4i*)gp, (LDS_AS v4i*)lp, 0, 0);
#else
            *(half8_t*)lp = *(const half8_t*)gp;
#endif
        }
    };

    auto compute = [&](int buf, int kk0) {
        const _Float16* lbase = &lds[buf * (GEMM_KC * GEMM_BN)];
        #pragma unroll
        for (int s = 0; s < GEMM_KC; s += 32) {
            const _Float16* ap = A + (size_t)arow * K + kk0 + s + kb;
            __builtin_prefetch(ap + GEMM_KC, 0, 1);       // global_prefetch_b8: next chunk's A
            union { half8_t h[2]; v16h v; } ua;
            ua.h[0] = *(const half8_t*)(ap);
            ua.h[1] = *(const half8_t*)(ap + 16);
            const _Float16* lb = lbase + (s + kb) * GEMM_BN + brow;
            union { half8_t h[2]; v16h v; } ub;
            #define WMMA_SUB(J, ACC)                                                     \
                for (int e = 0; e < 8; ++e) {                                            \
                    ub.v[e]     = lb[e * GEMM_BN + (J) * 16];                            \
                    ub.v[8 + e] = lb[(16 + e) * GEMM_BN + (J) * 16];                     \
                }                                                                        \
                ACC = __builtin_amdgcn_wmma_f32_16x16x32_f16(                            \
                          false, ua.v, false, ub.v, (short)0, ACC, false, false);
            WMMA_SUB(0, acc0)
            WMMA_SUB(1, acc1)
            WMMA_SUB(2, acc2)
            WMMA_SUB(3, acc3)
            #undef WMMA_SUB
        }
    };

    const int nch = K / GEMM_KC;
    stage(0, 0);
    for (int c = 0; c < nch; ++c) {
        if (c + 1 < nch) {
            stage((c + 1) & 1, (c + 1) * GEMM_KC);
#if USE_ASYNC_LDS
            __builtin_amdgcn_s_wait_asynccnt(2);   // chunk c landed; c+1 still in flight
#endif
        } else {
#if USE_ASYNC_LDS
            __builtin_amdgcn_s_wait_asynccnt(0);
#endif
        }
        __syncthreads();                           // chunk c visible to all waves
        compute(c & 1, c * GEMM_KC);
        __syncthreads();                           // all reads of chunk c done before restage
    }

    if (active) {
        // C/D layout: VGPR g, lanes 0-15 -> M=g, lanes 16-31 -> M=g+8; N = lane&15
        const int mbase = tm * 16 + ((lane >> 4) << 3);
        #pragma unroll
        for (int j = 0; j < 4; ++j) {
            v8f acc = (j == 0) ? acc0 : (j == 1) ? acc1 : (j == 2) ? acc2 : acc3;
            const int ncol = bcol0 + j * 16 + (lane & 15);
            const float badd = bias ? bias[ncol] : 0.f;
            #pragma unroll
            for (int g = 0; g < 8; ++g) {
                float v = acc[g] + badd;
                if (flags & GEMM_RELU) v = v > 0.f ? v : 0.f;
                size_t idx = (size_t)(mbase + g) * N + ncol;
                if (Cf) Cf[idx] = v;
                if (Ch) Ch[idx] = (_Float16)v;
            }
        }
    }
}

// ---------------------------------------------------------------------------
// Flash self-attention: one wave per (b, h, query); lane = head-dim channel.
// ---------------------------------------------------------------------------
__global__ void k_flash_attn(const float* __restrict__ Q, const float* __restrict__ Kk,
                             const float* __restrict__ V, _Float16* __restrict__ O16) {
    int wid = blockIdx.x * (blockDim.x >> 5) + (threadIdx.x >> 5);
    if (wid >= C_BS * C_NH * C_NQ) return;
    const int n = wid % C_NQ;
    const int t = wid / C_NQ;
    const int h = t % C_NH;
    const int b = t / C_NH;
    const int lane = threadIdx.x & 31;

    const size_t kvbase = (size_t)(b * C_NQ) * C_D + h * C_DH + lane;
    const float q = Q[(size_t)(b * C_NQ + n) * C_D + h * C_DH + lane];

    float mmax = -3.0e38f, l = 0.f, acc = 0.f;
    for (int m = 0; m < C_NQ; ++m) {
        float s = q * Kk[kvbase + (size_t)m * C_D];
        #pragma unroll
        for (int off = 16; off >= 1; off >>= 1) s += __shfl_xor(s, off, 32);
        s *= ATTN_SCALE;
        float nm   = fmaxf(mmax, s);
        float corr = __expf(mmax - nm);
        float p    = __expf(s - nm);
        acc = acc * corr + p * V[kvbase + (size_t)m * C_D];
        l   = l   * corr + p;
        mmax = nm;
    }
    O16[(size_t)(b * C_NQ + n) * C_D + h * C_DH + lane] = (_Float16)(acc / l);
}

// ---------------------------------------------------------------------------
// Fused residual + LayerNorm, one wave per row (256 cols, 8 per lane).
// ---------------------------------------------------------------------------
__global__ void k_resid_ln(const float* __restrict__ xin, const float* __restrict__ delta,
                           const float* __restrict__ gamma, const float* __restrict__ beta,
                           const float* __restrict__ qp,
                           float* __restrict__ xout, _Float16* __restrict__ x16,
                           _Float16* __restrict__ q16, int rows) {
    int row = blockIdx.x * (blockDim.x >> 5) + (threadIdx.x >> 5);
    if (row >= rows) return;
    const int lane = threadIdx.x & 31;
    const size_t base = (size_t)row * C_D;

    float v[8];
    float s = 0.f;
    #pragma unroll
    for (int i = 0; i < 8; ++i) {
        int c = i * 32 + lane;
        v[i] = xin[base + c] + delta[base + c];
        s += v[i];
    }
    #pragma unroll
    for (int off = 16; off >= 1; off >>= 1) s += __shfl_xor(s, off, 32);
    const float mean = s * (1.f / 256.f);

    float var = 0.f;
    #pragma unroll
    for (int i = 0; i < 8; ++i) { float d = v[i] - mean; var += d * d; }
    #pragma unroll
    for (int off = 16; off >= 1; off >>= 1) var += __shfl_xor(var, off, 32);
    const float inv = rsqrtf(var * (1.f / 256.f) + 1e-5f);

    #pragma unroll
    for (int i = 0; i < 8; ++i) {
        int c = i * 32 + lane;
        float y = (v[i] - mean) * inv * gamma[c] + beta[c];
        xout[base + c] = y;
        if (x16) x16[base + c] = (_Float16)y;
        if (q16) q16[base + c] = (_Float16)(y + qp[base + c]);
    }
}

// ---------------------------------------------------------------------------
// Softmax over 64 attention weights per (row, head); one wave, 2 elems/lane.
// ---------------------------------------------------------------------------
__global__ void k_softmax64(float* __restrict__ aw, int nWaves) {
    int wid = blockIdx.x * (blockDim.x >> 5) + (threadIdx.x >> 5);
    if (wid >= nWaves) return;
    const int lane = threadIdx.x & 31;
    const size_t base = (size_t)wid * 64;
    float a0 = aw[base + lane], a1 = aw[base + 32 + lane];
    float mx = fmaxf(a0, a1);
    #pragma unroll
    for (int off = 16; off >= 1; off >>= 1) mx = fmaxf(mx, __shfl_xor(mx, off, 32));
    float e0 = __expf(a0 - mx), e1 = __expf(a1 - mx);
    float s = e0 + e1;
    #pragma unroll
    for (int off = 16; off >= 1; off >>= 1) s += __shfl_xor(s, off, 32);
    float inv = 1.f / s;
    aw[base + lane] = e0 * inv;
    aw[base + 32 + lane] = e1 * inv;
}

// ---------------------------------------------------------------------------
// MS-deformable sampling: one wave per (b, n, h); lane = d channel.
// ---------------------------------------------------------------------------
__global__ void k_msdeform(const _Float16* __restrict__ value,
                           const float* __restrict__ off, const float* __restrict__ aw,
                           const float* __restrict__ ref, _Float16* __restrict__ out16) {
    int wid = blockIdx.x * (blockDim.x >> 5) + (threadIdx.x >> 5);
    if (wid >= C_BS * C_NQ * C_NH) return;
    const int h = wid % C_NH;
    const int t = wid / C_NH;
    const int n = t % C_NQ;
    const int b = t / C_NQ;
    const int lane = threadIdx.x & 31;

    const int HS[4] = {64, 32, 16, 8};
    const int ST[4] = {0, 4096, 5120, 5376};

    const size_t row = (size_t)(b * C_NQ + n);
    const float* offr = off + row * 1024 + h * 128;
    const float* awr  = aw  + row * 512  + h * 64;
    const float* refr = ref + row * 8;

    float acc = 0.f;
    for (int f = 0; f < C_NF; ++f) {
        const size_t vbase = ((size_t)(b * C_NF + f)) * C_S * C_D + h * C_DH + lane;
        for (int l = 0; l < C_NL; ++l) {
            const int HW = HS[l];
            const float invW = 1.f / (float)HW;
            const float rx = refr[l * 2 + 0], ry = refr[l * 2 + 1];
            for (int p = 0; p < C_NP; ++p) {
                const int oi = ((f * 4 + l) * 4 + p) * 2;
                const float w  = awr[(f * 4 + l) * 4 + p];
                const float xx = (rx + offr[oi + 0] * invW) * HW - 0.5f;
                const float yy = (ry + offr[oi + 1] * invW) * HW - 0.5f;
                const float x0 = floorf(xx), y0 = floorf(yy);
                float samp = 0.f;
                #pragma unroll
                for (int dy = 0; dy < 2; ++dy) {
                    #pragma unroll
                    for (int dx = 0; dx < 2; ++dx) {
                        float xi = x0 + dx, yi = y0 + dy;
                        if (xi >= 0.f && xi < (float)HW && yi >= 0.f && yi < (float)HW) {
                            float wgt = (1.f - fabsf(xx - xi)) * (1.f - fabsf(yy - yi));
                            int sidx = ST[l] + (int)yi * HW + (int)xi;
                            samp += wgt * (float)value[vbase + (size_t)sidx * C_D];
                        }
                    }
                }
                acc += w * samp;
            }
        }
    }
    out16[row * C_D + h * C_DH + lane] = (_Float16)acc;
}

// ---------------------------------------------------------------------------
// Host orchestration
// ---------------------------------------------------------------------------
static inline int wave_blocks(int nWaves) { return (nWaves + 7) / 8; }
static inline int gemm_blocks(int M, int N) { return ((M + GEMM_BM - 1) / GEMM_BM) * (N / GEMM_BN); }

extern "C" void kernel_launch(void* const* d_in, const int* in_sizes, int n_in,
                              void* d_out, int out_size, void* d_ws, size_t ws_size,
                              hipStream_t stream) {
    const float* tgt    = (const float*)d_in[0];
    const float* qpos   = (const float*)d_in[1];
    const float* refpts = (const float*)d_in[2];
    const float* src    = (const float*)d_in[3];
    const float* sa_wq = (const float*)d_in[6];  const float* sa_bq = (const float*)d_in[7];
    const float* sa_wk = (const float*)d_in[8];  const float* sa_bk = (const float*)d_in[9];
    const float* sa_wv = (const float*)d_in[10]; const float* sa_bv = (const float*)d_in[11];
    const float* sa_wo = (const float*)d_in[12]; const float* sa_bo = (const float*)d_in[13];
    const float* ca_wv = (const float*)d_in[14]; const float* ca_bv = (const float*)d_in[15];
    const float* ca_woff  = (const float*)d_in[16]; const float* ca_boff  = (const float*)d_in[17];
    const float* ca_wattn = (const float*)d_in[18]; const float* ca_battn = (const float*)d_in[19];
    const float* ca_wout  = (const float*)d_in[20]; const float* ca_bout  = (const float*)d_in[21];
    const float* ln1_g = (const float*)d_in[22]; const float* ln1_b = (const float*)d_in[23];
    const float* ln2_g = (const float*)d_in[24]; const float* ln2_b = (const float*)d_in[25];
    const float* ln3_g = (const float*)d_in[26]; const float* ln3_b = (const float*)d_in[27];
    const float* ffn_w1 = (const float*)d_in[28]; const float* ffn_b1 = (const float*)d_in[29];
    const float* ffn_w2 = (const float*)d_in[30]; const float* ffn_b2 = (const float*)d_in[31];
    float* out = (float*)d_out;

    // ---- workspace bump allocator (256B aligned) ----
    char* base = (char*)d_ws;
    size_t cur = 0;
    auto alloc = [&](size_t bytes) -> char* {
        char* p = base + cur;
        cur += (bytes + 255) & ~(size_t)255;
        return p;
    };
    _Float16* w_sa_wq = (_Float16*)alloc(65536 * 2);
    _Float16* w_sa_wk = (_Float16*)alloc(65536 * 2);
    _Float16* w_sa_wv = (_Float16*)alloc(65536 * 2);
    _Float16* w_sa_wo = (_Float16*)alloc(65536 * 2);
    _Float16* w_ca_wv = (_Float16*)alloc(65536 * 2);
    _Float16* w_ca_wout  = (_Float16*)alloc(65536 * 2);
    _Float16* w_ca_woff  = (_Float16*)alloc(262144 * 2);
    _Float16* w_ca_wattn = (_Float16*)alloc(131072 * 2);
    _Float16* w_ffn1 = (_Float16*)alloc(262144 * 2);
    _Float16* w_ffn2 = (_Float16*)alloc(262144 * 2);

    const size_t ND = (size_t)C_M1 * C_D;
    _Float16* q16 = (_Float16*)alloc(ND * 2);
    _Float16* x16 = (_Float16*)alloc(ND * 2);
    float*    x_cur = (float*)alloc(ND * 4);
    float*    qh = (float*)alloc(ND * 4);
    float*    kh = (float*)alloc(ND * 4);
    float*    vh = (float*)alloc(ND * 4);
    _Float16* o16 = (_Float16*)alloc(ND * 2);
    float*    proj = (float*)alloc(ND * 4);
    float*    offb = (float*)alloc((size_t)C_M1 * 1024 * 4);
    float*    awb  = (float*)alloc((size_t)C_M1 * 512 * 4);
    _Float16* src16   = (_Float16*)alloc((size_t)C_M2 * C_D * 2);
    _Float16* value16 = (_Float16*)alloc((size_t)C_M2 * C_D * 2);
    _Float16* samp16  = (_Float16*)alloc(ND * 2);
    _Float16* h16 = (_Float16*)alloc((size_t)C_M1 * C_DFFN * 2);

    auto conv = [&](const float* s, _Float16* d, int n) {
        int blocks = (n + 1023) / 1024;
        if (blocks > 4096) blocks = 4096;
        k_f32_to_f16<<<blocks, 256, 0, stream>>>(s, d, n);
    };
    conv(sa_wq, w_sa_wq, 65536);   conv(sa_wk, w_sa_wk, 65536);
    conv(sa_wv, w_sa_wv, 65536);   conv(sa_wo, w_sa_wo, 65536);
    conv(ca_wv, w_ca_wv, 65536);   conv(ca_wout, w_ca_wout, 65536);
    conv(ca_woff, w_ca_woff, 262144);
    conv(ca_wattn, w_ca_wattn, 131072);
    conv(ffn_w1, w_ffn1, 262144);  conv(ffn_w2, w_ffn2, 262144);
    conv(src, src16, C_M2 * C_D);

    k_prep<<<((int)ND + 255) / 256, 256, 0, stream>>>(tgt, qpos, q16, x16, (int)ND);

    // self-attention projections (WMMA)
    k_gemm_wmma<<<gemm_blocks(C_M1, C_D), 256, 0, stream>>>(q16, w_sa_wq, sa_bq, qh, nullptr, C_M1, C_D, C_D, 0);
    k_gemm_wmma<<<gemm_blocks(C_M1, C_D), 256, 0, stream>>>(q16, w_sa_wk, sa_bk, kh, nullptr, C_M1, C_D, C_D, 0);
    k_gemm_wmma<<<gemm_blocks(C_M1, C_D), 256, 0, stream>>>(x16, w_sa_wv, sa_bv, vh, nullptr, C_M1, C_D, C_D, 0);

    // flash attention
    k_flash_attn<<<wave_blocks(C_BS * C_NH * C_NQ), 256, 0, stream>>>(qh, kh, vh, o16);

    // output projection + residual + LN2 (emit cq16 into q16)
    k_gemm_wmma<<<gemm_blocks(C_M1, C_D), 256, 0, stream>>>(o16, w_sa_wo, sa_bo, proj, nullptr, C_M1, C_D, C_D, 0);
    k_resid_ln<<<wave_blocks(C_M1), 256, 0, stream>>>(tgt, proj, ln2_g, ln2_b, qpos, x_cur, x16, q16, C_M1);

    // cross-attention: value projection (f16 out), offsets, attn weights
    k_gemm_wmma<<<gemm_blocks(C_M2, C_D), 256, 0, stream>>>(src16, w_ca_wv, ca_bv, nullptr, value16, C_M2, C_D, C_D, 0);
    k_gemm_wmma<<<gemm_blocks(C_M1, 1024), 256, 0, stream>>>(q16, w_ca_woff, ca_boff, offb, nullptr, C_M1, C_D, 1024, 0);
    k_gemm_wmma<<<gemm_blocks(C_M1, 512), 256, 0, stream>>>(q16, w_ca_wattn, ca_battn, awb, nullptr, C_M1, C_D, 512, 0);
    k_softmax64<<<wave_blocks(C_M1 * C_NH), 256, 0, stream>>>(awb, C_M1 * C_NH);

    // deformable sampling + output projection + residual + LN1
    k_msdeform<<<wave_blocks(C_BS * C_NQ * C_NH), 256, 0, stream>>>(value16, offb, awb, refpts, samp16);
    k_gemm_wmma<<<gemm_blocks(C_M1, C_D), 256, 0, stream>>>(samp16, w_ca_wout, ca_bout, proj, nullptr, C_M1, C_D, C_D, 0);
    k_resid_ln<<<wave_blocks(C_M1), 256, 0, stream>>>(x_cur, proj, ln1_g, ln1_b, nullptr, x_cur, x16, nullptr, C_M1);

    // FFN + residual + LN3 -> d_out
    k_gemm_wmma<<<gemm_blocks(C_M1, C_DFFN), 256, 0, stream>>>(x16, w_ffn1, ffn_b1, nullptr, h16, C_M1, C_D, C_DFFN, GEMM_RELU);
    k_gemm_wmma<<<gemm_blocks(C_M1, C_D), 256, 0, stream>>>(h16, w_ffn2, ffn_b2, proj, nullptr, C_M1, C_DFFN, C_D, 0);
    k_resid_ln<<<wave_blocks(C_M1), 256, 0, stream>>>(x_cur, proj, ln3_g, ln3_b, nullptr, out, nullptr, nullptr, C_M1);

    (void)in_sizes; (void)n_in; (void)out_size; (void)ws_size;
}
